// Attention_10694468567284
// MI455X (gfx1250) — compile-verified, matrix operand order checked by reading, not run on hardware
//
#include <hip/hip_runtime.h>

// ---------------------------------------------------------------------------
// Flash-attention for B=16, S=2048, D=128 (fp32 in/out, bool mask) on gfx1250.
// bf16 WMMA (v_wmma_f32_16x16x32_bf16) for QK^T and PV, fp32 online softmax.
// Double-buffered LDS (one barrier/iter): K, V^T (4x4 in-reg transpose), and
// the boolean mask (repacked per-lane via v_perm at staging time).
// 2 workgroups per WGP (248KB of 320KB LDS) for cross-WG latency hiding.
// ---------------------------------------------------------------------------

typedef __attribute__((ext_vector_type(16))) __bf16 v16bf;
typedef __attribute__((ext_vector_type(8)))  __bf16 v8bf;
typedef __attribute__((ext_vector_type(8)))  float  v8f;
typedef __attribute__((ext_vector_type(4)))  float  v4f;
typedef __attribute__((ext_vector_type(2)))  unsigned int u32x2;
typedef __attribute__((ext_vector_type(4)))  unsigned int u32x4;

#define SQ   2048
#define DD   128
#define BR   128      // q rows per workgroup (16 per wave x 8 waves)
#define BC   64       // kv columns per iteration
#define NKB  (SQ / BC)
#define KSTR 136      // LDS row stride (elems) for K tile
#define VSTR 72       // LDS row stride (elems) for V^T tile (feature-major)
#define PSTR 72       // LDS row stride (elems) for P tile
#define MSTR 72       // LDS row stride (ushorts) for packed mask rows (144B)

#define SK_ELEMS  (BC * KSTR)               // 8704
#define SV_ELEMS  (DD * VSTR)               // 9216
#define SM_ELEMS  (BR * MSTR)               // 9216
#define SV_BASE   (2 * SK_ELEMS)            // 17408
#define SM_BASE   (SV_BASE + 2 * SV_ELEMS)  // 35840
#define SP_BASE   (SM_BASE + 2 * SM_ELEMS)  // 54272
#define SMEM_TOT  (SP_BASE + 8 * 16 * PSTR) // 63488 ushorts = 126976 B

// Pack two fp32 -> two bf16 (round-to-nearest-away) in one v_perm_b32.
__device__ __forceinline__ unsigned pack_bf16(float a, float b) {
  unsigned ua = __builtin_bit_cast(unsigned, a) + 0x8000u;
  unsigned ub = __builtin_bit_cast(unsigned, b) + 0x8000u;
  return __builtin_amdgcn_perm(ub, ua, 0x07060302u);
}
__device__ __forceinline__ unsigned short bf16_1(float f) {
  return (unsigned short)((__builtin_bit_cast(unsigned, f) + 0x8000u) >> 16);
}

__device__ __forceinline__ v8f wmma_bf16(v16bf a, v16bf b, v8f c) {
  return __builtin_amdgcn_wmma_f32_16x16x32_bf16(false, a, false, b, (short)0, c,
                                                 false, false);
}

__device__ __forceinline__ v16bf lds_frag(const unsigned short* p) {
  v8bf lo = *(const v8bf*)(p);
  v8bf hi = *(const v8bf*)(p + 16);
  return __builtin_shufflevector(lo, hi, 0,1,2,3,4,5,6,7,8,9,10,11,12,13,14,15);
}
__device__ __forceinline__ v16bf lds_frag2(const unsigned short* p0,
                                           const unsigned short* p1) {
  v8bf lo = *(const v8bf*)(p0);
  v8bf hi = *(const v8bf*)(p1);
  return __builtin_shufflevector(lo, hi, 0,1,2,3,4,5,6,7,8,9,10,11,12,13,14,15);
}

// ---- tile staging helpers --------------------------------------------------
__device__ __forceinline__ void load_tileK(const float* kt, int tid, v4f* kreg) {
  #pragma unroll
  for (int i = 0; i < 8; ++i) {
    const int f = i * 256 + tid, r = f >> 5, c4 = f & 31;
    kreg[i] = *(const v4f*)(kt + r * DD + c4 * 4);
  }
}
__device__ __forceinline__ void stage_tileK(const v4f* kreg, unsigned short* sK, int tid) {
  #pragma unroll
  for (int i = 0; i < 8; ++i) {
    const int f = i * 256 + tid, r = f >> 5, c4 = f & 31;
    u32x2 pk;
    pk.x = pack_bf16(kreg[i][0], kreg[i][1]);
    pk.y = pack_bf16(kreg[i][2], kreg[i][3]);
    *(u32x2*)&sK[r * KSTR + c4 * 4] = pk;   // conflict-free b64 stores
  }
}
// V: each thread owns two 4x4 blocks (4 kv rows x 4 features).
__device__ __forceinline__ void load_tileV(const float* vt, int tid, v4f* vreg) {
  #pragma unroll
  for (int blk = 0; blk < 2; ++blk) {
    const int bk = tid + 256 * blk;
    const int c4 = bk & 31;            // feature float4 column
    const int r0 = (bk >> 5) * 4;      // kv row base
    #pragma unroll
    for (int rr = 0; rr < 4; ++rr)
      vreg[4 * blk + rr] = *(const v4f*)(vt + (r0 + rr) * DD + c4 * 4);
  }
}
// Feature-major V^T with kv-group-of-4 XOR swizzle: elem (d, kv) at
//   d*VSTR + (((kv>>2) ^ (2*((d>>4)&3))) << 2) + (kv&3)
__device__ __forceinline__ void stage_tileV(const v4f* vreg, unsigned short* sV, int tid) {
  #pragma unroll
  for (int blk = 0; blk < 2; ++blk) {
    const int bk = tid + 256 * blk;
    const int c4 = bk & 31;
    const int r0 = (bk >> 5) * 4;
    const int g4 = r0 >> 2;
    #pragma unroll
    for (int j = 0; j < 4; ++j) {
      const int d = 4 * c4 + j;
      const int x = 2 * ((d >> 4) & 3);
      u32x2 w;                                   // in-register 4x4 transpose
      w.x = pack_bf16(vreg[4 * blk + 0][j], vreg[4 * blk + 1][j]);
      w.y = pack_bf16(vreg[4 * blk + 2][j], vreg[4 * blk + 3][j]);
      *(u32x2*)&sV[d * VSTR + ((g4 ^ x) << 2)] = w;   // b64 stores
    }
  }
}
// Mask: 2 threads per q row; thread (r, w) loads row r's cols 32w..32w+31.
__device__ __forceinline__ void load_tileM(const unsigned char* mtile, int tid,
                                           u32x4* mreg) {
  const int r = tid >> 1, w = tid & 1;
  const unsigned char* p = mtile + (size_t)r * SQ + 32 * w;
  mreg[0] = __builtin_nontemporal_load((const u32x4*)p);
  mreg[1] = __builtin_nontemporal_load((const u32x4*)(p + 16));
}
// Repack so lane n reads one b64: entry (row, n) = u16x4 {m(n), m(n+16), m(n+32), m(n+48)}.
// Thread (r, w=0) writes halves {n, n+16}; (r, w=1) writes {n+32, n+48}.
__device__ __forceinline__ void stage_tileM(const u32x4* mreg, unsigned short* sM, int tid) {
  const int r = tid >> 1, w = tid & 1;
  unsigned m[8] = { mreg[0][0], mreg[0][1], mreg[0][2], mreg[0][3],
                    mreg[1][0], mreg[1][1], mreg[1][2], mreg[1][3] };
  #pragma unroll
  for (int nn = 0; nn < 16; ++nn) {
    const int i = nn >> 2, j = nn & 3;
    // bytes [j, j, 4+j, 4+j] -> mask to [b(n),0,b(n+16),0] = two u16s
    const unsigned sel = (unsigned)j * 0x00000101u + (unsigned)(4 + j) * 0x01010000u;
    unsigned pr = __builtin_amdgcn_perm(m[i + 4], m[i], sel) & 0x00FF00FFu;
    *(unsigned*)&sM[r * MSTR + nn * 4 + 2 * w] = pr;
  }
}

__launch_bounds__(256, 2)
__global__ void fa_fwd_kernel(const float* __restrict__ Q,
                              const float* __restrict__ K,
                              const float* __restrict__ V,
                              const unsigned char* __restrict__ Msk,
                              float* __restrict__ Out)
{
  extern __shared__ __align__(16) unsigned short smem[];

  const int tid  = threadIdx.x;
  const int wave = tid >> 5;
  const int lane = tid & 31;
  const int n    = lane & 15;
  const int h    = lane >> 4;

  const int nqb   = SQ / BR;
  const int b     = blockIdx.x / nqb;
  const int qblk  = blockIdx.x % nqb;
  const int qbase = qblk * BR + wave * 16;

  const float QSCALE = 0.08838834764831845f * 1.4426950408889634f; // 1/sqrt(128)*log2(e)

  // ---- Q strip (16 x 128) -> 4 bf16 A-fragments, pre-scaled -----------------
  v16bf qa[4];
  {
    const float* qrow = Q + ((size_t)b * SQ + qbase + n) * DD;
    #pragma unroll
    for (int kk = 0; kk < 4; ++kk) {
      union { unsigned u[8]; v16bf v; } cvt;
      #pragma unroll
      for (int half = 0; half < 2; ++half) {
        const int f0 = 32 * kk + 16 * half + 8 * h;
        v4f a0 = *(const v4f*)(qrow + f0);
        v4f a1 = *(const v4f*)(qrow + f0 + 4);
        cvt.u[4 * half + 0] = pack_bf16(a0[0] * QSCALE, a0[1] * QSCALE);
        cvt.u[4 * half + 1] = pack_bf16(a0[2] * QSCALE, a0[3] * QSCALE);
        cvt.u[4 * half + 2] = pack_bf16(a1[0] * QSCALE, a1[1] * QSCALE);
        cvt.u[4 * half + 3] = pack_bf16(a1[2] * QSCALE, a1[3] * QSCALE);
      }
      qa[kk] = cvt.v;
    }
  }

  const v8f vzero = {0.f, 0.f, 0.f, 0.f, 0.f, 0.f, 0.f, 0.f};
  v8f o[8];
  #pragma unroll
  for (int d = 0; d < 8; ++d) o[d] = vzero;
  float mrun[8], lrun[8];
  #pragma unroll
  for (int v = 0; v < 8; ++v) { mrun[v] = -3.0e38f; lrun[v] = 0.0f; }

  unsigned short* sPw = smem + SP_BASE + wave * 16 * PSTR;
  const float* kbase = K + (size_t)b * SQ * DD;
  const float* vbase = V + (size_t)b * SQ * DD;
  const unsigned char* mtile0 = Msk + (size_t)b * SQ * SQ + (size_t)(qblk * BR) * SQ;

  // ---- prologue: stage tile 0 into buffer 0 --------------------------------
  {
    v4f kreg[8];
    load_tileK(kbase, tid, kreg);
    stage_tileK(kreg, smem, tid);
    u32x4 mreg[2];
    load_tileM(mtile0, tid, mreg);
    stage_tileM(mreg, smem + SM_BASE, tid);
    v4f vreg[8];
    load_tileV(vbase, tid, vreg);
    stage_tileV(vreg, smem + SV_BASE, tid);
  }

  for (int kb = 0; kb < NKB; ++kb) {
    const int cur = kb & 1;
    const unsigned short* sKc = smem + cur * SK_ELEMS;
    const unsigned short* sVc = smem + SV_BASE + cur * SV_ELEMS;
    const unsigned short* sMc = smem + SM_BASE + cur * SM_ELEMS;
    unsigned short* sKn = smem + (cur ^ 1) * SK_ELEMS;
    unsigned short* sVn = smem + SV_BASE + (cur ^ 1) * SV_ELEMS;
    unsigned short* sMn = smem + SM_BASE + (cur ^ 1) * SM_ELEMS;
    const bool have_next = (kb + 1) < NKB;

    // next tile's K + mask loads in flight across the barrier (~40 VGPRs)
    v4f kreg[8];
    u32x4 mreg[2];
    if (have_next) {
      load_tileK(kbase + (size_t)(kb + 1) * BC * DD, tid, kreg);
      load_tileM(mtile0 + (kb + 1) * BC, tid, mreg);
    }
    if (kb + 2 < NKB) {
      __builtin_prefetch(kbase + (size_t)(kb + 2) * BC * DD + tid * 32, 0, 3);
      __builtin_prefetch(vbase + (size_t)(kb + 2) * BC * DD + tid * 32, 0, 3);
    }

    __syncthreads();   // single barrier per iteration (ping-pong)

    if (have_next) {
      stage_tileK(kreg, sKn, tid);          // kreg dies here
      stage_tileM(mreg, sMn, tid);
      v4f vreg[8];
      load_tileV(vbase + (size_t)(kb + 1) * BC * DD, tid, vreg);
      stage_tileV(vreg, sVn, tid);          // V latency overlaps compute below
    }

    // ---- S = (Q*scale) K^T -------------------------------------------------
    v8f s[4];
    #pragma unroll
    for (int c = 0; c < 4; ++c) s[c] = vzero;
    #pragma unroll
    for (int kk = 0; kk < 4; ++kk) {
      v16bf bb[4];
      #pragma unroll
      for (int c = 0; c < 4; ++c)
        bb[c] = lds_frag(&sKc[(16 * c + n) * KSTR + 32 * kk + 8 * h]);
      #pragma unroll
      for (int c = 0; c < 4; ++c)
        s[c] = wmma_bf16(qa[kk], bb[c], s[c]);
    }

    // ---- fused per-row: mask, online softmax, P->LDS, O rescale ------------
    #pragma unroll
    for (int v = 0; v < 8; ++v) {
      const int mrow = (wave << 4) + v + 8 * h;   // q row within 128-row tile
      u32x2 mw = *(const u32x2*)&sMc[mrow * MSTR + n * 4];
      float sv[4];
      sv[0] = (unsigned short)(mw.x)       ? -__builtin_inff() : s[0][v];
      sv[1] = (mw.x >> 16)                 ? -__builtin_inff() : s[1][v];
      sv[2] = (unsigned short)(mw.y)       ? -__builtin_inff() : s[2][v];
      sv[3] = (mw.y >> 16)                 ? -__builtin_inff() : s[3][v];
      float rm = fmaxf(fmaxf(sv[0], sv[1]), fmaxf(sv[2], sv[3]));
      rm = fmaxf(rm, __shfl_xor(rm, 1));
      rm = fmaxf(rm, __shfl_xor(rm, 2));
      rm = fmaxf(rm, __shfl_xor(rm, 4));
      rm = fmaxf(rm, __shfl_xor(rm, 8));
      const float mo = mrun[v];
      const float mn = fmaxf(mo, rm);
      const float cf = __builtin_amdgcn_exp2f(mo - mn);
      mrun[v] = mn;
      float lacc = lrun[v] * cf;
      #pragma unroll
      for (int c = 0; c < 4; ++c) {
        float p = __builtin_amdgcn_exp2f(sv[c] - mn);
        lacc += p;
        sPw[(v + 8 * h) * PSTR + 16 * c + n] = bf16_1(p);
      }
      lrun[v] = lacc;
      #pragma unroll
      for (int d = 0; d < 8; ++d) o[d][v] *= cf;
    }

    // ---- O += P V (swizzled V^T fragment loads) ----------------------------
    #pragma unroll
    for (int kk = 0; kk < 2; ++kk) {
      v16bf pa = lds_frag(&sPw[n * PSTR + 32 * kk + 8 * h]);
      const int g4a = 8 * kk + 2 * h;
      #pragma unroll
      for (int dg = 0; dg < 2; ++dg) {
        v16bf vb[4];
        #pragma unroll
        for (int j = 0; j < 4; ++j) {
          const int dd = 4 * dg + j;            // feature 16-chunk 0..7
          const int dfeat = 16 * dd + n;
          const int x = 2 * (dd & 3);
          vb[j] = lds_frag2(&sVc[dfeat * VSTR + (((g4a    ) ^ x) << 2)],
                            &sVc[dfeat * VSTR + (((g4a + 4) ^ x) << 2)]);
        }
        #pragma unroll
        for (int j = 0; j < 4; ++j)
          o[4 * dg + j] = wmma_bf16(pa, vb[j], o[4 * dg + j]);
      }
    }
  }

  // ---- finalize ------------------------------------------------------------
  #pragma unroll
  for (int v = 0; v < 8; ++v) {
    float l = lrun[v];
    l += __shfl_xor(l, 1);
    l += __shfl_xor(l, 2);
    l += __shfl_xor(l, 4);
    l += __shfl_xor(l, 8);
    lrun[v] = 1.0f / l;
  }
  float* orow = Out + ((size_t)b * SQ + qbase) * DD;
  #pragma unroll
  for (int d = 0; d < 8; ++d) {
    #pragma unroll
    for (int v = 0; v < 8; ++v) {
      __builtin_nontemporal_store(o[d][v] * lrun[v],
                                  orow + (v + 8 * h) * DD + 16 * d + n);
    }
  }
}

extern "C" void kernel_launch(void* const* d_in, const int* in_sizes, int n_in,
                              void* d_out, int out_size, void* d_ws, size_t ws_size,
                              hipStream_t stream) {
  const float* Q = (const float*)d_in[0];
  const float* K = (const float*)d_in[1];
  const float* V = (const float*)d_in[2];
  const unsigned char* M = (const unsigned char*)d_in[3];
  float* O = (float*)d_out;
  (void)in_sizes; (void)n_in; (void)out_size; (void)d_ws; (void)ws_size;

  dim3 grid(16 * (SQ / BR));   // 256 workgroups
  dim3 block(256);             // 8 wave32s
  size_t shmem = (size_t)SMEM_TOT * sizeof(unsigned short);  // 126976 B
  hipLaunchKernelGGL(fa_fwd_kernel, grid, block, shmem, stream, Q, K, V, M, O);
}
